// HGNN_24068996727339
// MI455X (gfx1250) — matrix operand order repaired and checked by reading, not compiled, not run on hardware
//
#include <hip/hip_runtime.h>

// ---------------- problem constants (from reference) ----------------
#define N_NODES  10000
#define N_EDGES  160000
#define N_GRAPHS 256
#define HD       512          // HEADS*HIDDEN
#define NPAD     10112        // 79*128, GEMM M-tile multiple
#define K0PAD    64           // embed dim 39 padded to 64
#define ENC_NEG_INF 0x007FFFFFu

#define USE_ASYNC_A 1         // async global->LDS staging for A tiles (CDNA5 ASYNCcnt path)

typedef __attribute__((ext_vector_type(16))) __bf16 v16bf;
typedef __attribute__((ext_vector_type(8)))  float  v8f;

// f32 -> bf16 round-to-nearest-even
static __device__ __forceinline__ unsigned short f2bf(float f) {
  unsigned u = __float_as_uint(f);
  u += 0x7FFFu + ((u >> 16) & 1u);
  return (unsigned short)(u >> 16);
}
// monotone float<->uint map for atomicMax-based segment_max
static __device__ __forceinline__ unsigned encf(float f) {
  unsigned u = __float_as_uint(f);
  return (u & 0x80000000u) ? ~u : (u | 0x80000000u);
}
static __device__ __forceinline__ float decf(unsigned u) {
  return __uint_as_float((u & 0x80000000u) ? (u & 0x7FFFFFFFu) : ~u);
}

// ---------------- utility fill ----------------
__global__ void fill_u32(unsigned* __restrict__ p, unsigned v, long n) {
  long i = (long)blockIdx.x * 256 + threadIdx.x;
  if (i < n) p[i] = v;
}

// ---------------- embed: one-hot(8 cat) + 8 cont -> bf16 [NPAD,64] ----------------
__global__ void embed_kernel(const float* __restrict__ x, unsigned short* __restrict__ embb) {
  int n = blockIdx.x * 256 + threadIdx.x;
  if (n >= NPAD) return;
  unsigned short* row = embb + (long)n * K0PAD;
  #pragma unroll
  for (int j = 0; j < K0PAD; ++j) row[j] = 0;
  if (n < N_NODES) {
    const int offs[8] = {0, 7, 12, 16, 20, 22, 24, 28};
    #pragma unroll
    for (int i = 0; i < 8; ++i) {
      int v = (int)x[n * 16 + i];
      row[offs[i] + v] = 0x3F80;  // bf16(1.0)
    }
    #pragma unroll
    for (int j = 0; j < 8; ++j) row[31 + j] = f2bf(x[n * 16 + 8 + j]);
  }
}

// ---- weights: f32 [Ksrc,512] -> bf16 TRANSPOSED [512,Kdst] (zero-padded K) ----
__global__ void conv_wt_kernel(const float* __restrict__ src, unsigned short* __restrict__ dst,
                               int Ksrc, int Kdst) {
  int i = blockIdx.x * 256 + threadIdx.x;  // over 512*Kdst
  if (i >= 512 * Kdst) return;
  int c = i / Kdst, k = i - c * Kdst;
  dst[i] = (k < Ksrc) ? f2bf(src[k * 512 + c]) : (unsigned short)0;
}

// ---------------- bf16 WMMA GEMM: C[M,512] = A[M,K](bf16) * Wt[512,K](bf16)^T ----------------
// block = 256 threads (8 wave32), block tile 128M x 128N, wave tile 32M x 64N (2x4 frags).
// A staged in LDS via async global->LDS copy; B fragments loaded directly from
// transposed global weights (L2-resident). K stepped by 64.
#define KSTEP 64
#define LDSA  72   // ushort stride: 36-dword row pitch -> conflict-free 16-lane b128 reads
__global__ __launch_bounds__(256)
void gemm_bf16_wmma(const unsigned short* __restrict__ Abf,  // [M,K] bf16
                    const unsigned short* __restrict__ Wt,   // [512,K] bf16 (N-major)
                    float* __restrict__ C,                   // [M,512] f32
                    int M, int K) {
  __shared__ unsigned short As[128 * LDSA];
  const int tid  = threadIdx.x;
  const int lane = tid & 31, warp = tid >> 5;
  const int wm = warp >> 1, wn = warp & 1;      // 4x2 wave grid -> 32M x 64N per wave
  const int r16 = lane & 15, hi = lane >> 4;
  const int bM = blockIdx.y * 128, bN = blockIdx.x * 128;

  v8f acc[2][4];
  #pragma unroll
  for (int m = 0; m < 2; ++m)
    #pragma unroll
    for (int n = 0; n < 4; ++n) acc[m][n] = (v8f)0.0f;

  for (int k0 = 0; k0 < K; k0 += KSTEP) {
    __syncthreads();
    // stage A tile 128x64 bf16: each thread copies 4x 16B chunks (coalesced 128B per 8 lanes)
    #pragma unroll
    for (int c = 0; c < 4; ++c) {
      int idx = c * 2048 + tid * 8;
      int r = idx >> 6, k = idx & 63;
      const unsigned short* gp = Abf + (bM + r) * K + k0 + k;
      unsigned short* lp = &As[r * LDSA + k];
#if USE_ASYNC_A
      unsigned lp32 = (unsigned)(unsigned long long)lp;  // low 32 bits = LDS byte offset
      asm volatile("global_load_async_to_lds_b128 %0, %1, off"
                   :: "v"(lp32), "v"((const void*)gp) : "memory");
#else
      *(uint4*)lp = *(const uint4*)gp;
#endif
    }
#if USE_ASYNC_A
    asm volatile("s_wait_asynccnt 0x0" ::: "memory");
#endif
    __syncthreads();

    #pragma unroll
    for (int ks = 0; ks < KSTEP; ks += 32) {
      union UB { unsigned u[8]; v16bf v; };
      UB a[2], b[4];
      // A fragments from LDS in contiguous k order (k = 16*hi + s), same as B
      #pragma unroll
      for (int m = 0; m < 2; ++m) {
        const unsigned short* ap = &As[(wm * 32 + m * 16 + r16) * LDSA + ks];
        int kb = hi * 16;  // lanes 0-15: K 0..15 ; lanes 16-31: K 16..31
        #pragma unroll
        for (int j = 0; j < 4; ++j) a[m].u[j]     = *(const unsigned*)(ap + kb + 2 * j);
        #pragma unroll
        for (int j = 0; j < 4; ++j) a[m].u[4 + j] = *(const unsigned*)(ap + kb + 8 + 2 * j);
      }
      // B fragments straight from global transposed weights (16 contiguous bf16 per lane)
      #pragma unroll
      for (int n = 0; n < 4; ++n) {
        const unsigned short* bp = Wt + (bN + wn * 64 + n * 16 + r16) * K + k0 + ks + hi * 16;
        uint4 lo  = *(const uint4*)bp;
        uint4 hi4 = *(const uint4*)(bp + 8);
        b[n].u[0] = lo.x;  b[n].u[1] = lo.y;  b[n].u[2] = lo.z;  b[n].u[3] = lo.w;
        b[n].u[4] = hi4.x; b[n].u[5] = hi4.y; b[n].u[6] = hi4.z; b[n].u[7] = hi4.w;
      }
      #pragma unroll
      for (int m = 0; m < 2; ++m)
        #pragma unroll
        for (int n = 0; n < 4; ++n)
          acc[m][n] = __builtin_amdgcn_wmma_f32_16x16x32_bf16(
              false, a[m].v, false, b[n].v, (short)0, acc[m][n], false, false);
    }
  }

  // C layout: lane%16 = N col, VGPR r = M row (+8 for upper half-wave).
  // N==512 compile-time -> base pointer + immediate offsets.
  #pragma unroll
  for (int m = 0; m < 2; ++m)
    #pragma unroll
    for (int n = 0; n < 4; ++n) {
      float* cp = C + (bM + wm * 32 + m * 16 + hi * 8) * 512
                    + (bN + wn * 64 + n * 16 + r16);
      #pragma unroll
      for (int r = 0; r < 8; ++r) cp[r * 512] = acc[m][n][r];
    }
}

// ---------------- attention scores e_s/e_d: one wave per node ----------------
__global__ __launch_bounds__(256)
void scores_kernel(const float* __restrict__ h, const float* __restrict__ a_src,
                   const float* __restrict__ a_dst, float* __restrict__ es,
                   float* __restrict__ ed) {
  int node = blockIdx.x * 8 + (threadIdx.x >> 5);
  if (node >= N_NODES) return;
  int lane = threadIdx.x & 31;
  const float* hp = h + (long)node * HD + lane * 16;
  float vs = 0.f, vd = 0.f;
  #pragma unroll
  for (int t = 0; t < 16; ++t) {
    float hv = hp[t];
    vs += hv * a_src[lane * 16 + t];  // flat [H,F] index == feature column
    vd += hv * a_dst[lane * 16 + t];
  }
  #pragma unroll
  for (int off = 1; off < 8; off <<= 1) {   // reduce within 8-lane head group
    vs += __shfl_xor(vs, off, 32);
    vd += __shfl_xor(vd, off, 32);
  }
  if ((lane & 7) == 0) {
    int head = lane >> 3;
    es[node * 4 + head] = vs;
    ed[node * 4 + head] = vd;
  }
}

// ---------------- edge logits + segment_max(dst) ----------------
__global__ void edge_logit_max(const float* __restrict__ es, const float* __restrict__ ed,
                               const int* __restrict__ src, const int* __restrict__ dst,
                               float* __restrict__ ew, unsigned* __restrict__ menc) {
  int idx = blockIdx.x * 256 + threadIdx.x;
  if (idx >= N_EDGES * 4) return;
  int e = idx >> 2, hch = idx & 3;
  float v = es[src[e] * 4 + hch] + ed[dst[e] * 4 + hch];
  v = v > 0.f ? v : 0.01f * v;   // leaky_relu(alpha=0.01)
  ew[idx] = v;
  atomicMax(&menc[dst[e] * 4 + hch], encf(v));
}

// ---------------- w = exp(e - m[dst]) and segment_sum(dst) ----------------
__global__ void edge_w_denom(const int* __restrict__ dst, float* __restrict__ ew,
                             const unsigned* __restrict__ menc, float* __restrict__ denom) {
  int idx = blockIdx.x * 256 + threadIdx.x;
  if (idx >= N_EDGES * 4) return;
  int e = idx >> 2, hch = idx & 3;
  int d4 = dst[e] * 4 + hch;
  float w = __expf(ew[idx] - decf(menc[d4]));
  ew[idx] = w;
  atomicAdd(&denom[d4], w);
}

// ---------------- out[dst] += att * h[src] : one wave per edge ----------------
__global__ __launch_bounds__(256)
void aggregate_kernel(const float* __restrict__ h, const float* __restrict__ ew,
                      const float* __restrict__ denom, const int* __restrict__ src,
                      const int* __restrict__ dst, float* __restrict__ out) {
  int e = blockIdx.x * 8 + (threadIdx.x >> 5);
  if (e >= N_EDGES) return;
  int lane = threadIdx.x & 31;
  int s = src[e], d = dst[e];
  int head = lane >> 3;                 // 16 contiguous feats per lane stay in one head
  float att = ew[e * 4 + head] / (denom[d * 4 + head] + 1e-16f);
  const float4* hs = (const float4*)(h + (long)s * HD + lane * 16);  // 64B aligned
  float* od = out + (long)d * HD + lane * 16;
  #pragma unroll
  for (int q = 0; q < 4; ++q) {
    float4 v = hs[q];
    atomicAdd(&od[q * 4 + 0], att * v.x);
    atomicAdd(&od[q * 4 + 1], att * v.y);
    atomicAdd(&od[q * 4 + 2], att * v.z);
    atomicAdd(&od[q * 4 + 3], att * v.w);
  }
}

// ---- elu (+f32/bf16 outputs, optional residual x3 = resid + elu_out) ----
__global__ void elu_res_kernel(const float* __restrict__ agg, float* __restrict__ out,
                               unsigned short* __restrict__ out_bf,
                               const float* __restrict__ resid, float* __restrict__ out2,
                               unsigned short* __restrict__ out2_bf, long n) {
  long i = (long)blockIdx.x * 256 + threadIdx.x;
  if (i >= n) return;
  float v = agg[i];
  v = v > 0.f ? v : (__expf(v) - 1.f);
  out[i] = v;
  if (out_bf) out_bf[i] = f2bf(v);
  if (resid) {
    float r = resid[i] + v;
    out2[i] = r;
    if (out2_bf) out2_bf[i] = f2bf(r);
  }
}

// ---------------- concat + segment_max pooling over batch ----------------
__global__ void pool_max_kernel(const float* __restrict__ xt, const float* __restrict__ x1,
                                const float* __restrict__ x2, const float* __restrict__ x3,
                                const int* __restrict__ batch, unsigned* __restrict__ penc) {
  long i = (long)blockIdx.x * 256 + threadIdx.x;
  if (i >= (long)N_NODES * 2048) return;
  int node = (int)(i >> 11), f = (int)(i & 2047);
  const float* s = (f & 1024) ? ((f & 512) ? x3 : x2) : ((f & 512) ? x1 : xt);
  float v = s[(long)node * HD + (f & 511)];
  atomicMax(&penc[batch[node] * 2048 + f], encf(v));
}

__global__ void pool_dec_kernel(const unsigned* __restrict__ penc, float* __restrict__ pf,
                                unsigned short* __restrict__ pbf) {
  int i = blockIdx.x * 256 + threadIdx.x;
  if (i >= N_GRAPHS * 2048) return;
  float v = decf(penc[i]);
  pf[i] = v;
  pbf[i] = f2bf(v);
}

// ---------------- batchnorm over graphs + relu (one thread per column) ----------------
__global__ void bn_kernel(float* __restrict__ y, const float* __restrict__ b1,
                          const float* __restrict__ gamma, const float* __restrict__ beta) {
  int c = threadIdx.x;  // 512 threads
  float s = 0.f, s2 = 0.f;
  for (int g = 0; g < N_GRAPHS; ++g) {
    float v = y[g * HD + c] + b1[c];
    s += v; s2 += v * v;
  }
  float mu = s * (1.f / N_GRAPHS);
  float var = s2 * (1.f / N_GRAPHS) - mu * mu;
  float inv = rsqrtf(var + 1e-5f) * gamma[c];
  for (int g = 0; g < N_GRAPHS; ++g) {
    float v = (y[g * HD + c] + b1[c] - mu) * inv + beta[c];
    y[g * HD + c] = fmaxf(v, 0.f);
  }
}

// ---------------- final matvec: out[g] = relu_y[g] . w2 + b2 ----------------
__global__ __launch_bounds__(256)
void final_kernel(const float* __restrict__ y, const float* __restrict__ w2,
                  const float* __restrict__ b2, float* __restrict__ out) {
  int g = blockIdx.x * 8 + (threadIdx.x >> 5);
  if (g >= N_GRAPHS) return;
  int lane = threadIdx.x & 31;
  float s = 0.f;
  #pragma unroll
  for (int t = 0; t < 16; ++t) s += y[g * HD + lane * 16 + t] * w2[lane * 16 + t];
  #pragma unroll
  for (int off = 16; off >= 1; off >>= 1) s += __shfl_xor(s, off, 32);
  if (lane == 0) out[g] = s + b2[0];
}

// =====================================================================
extern "C" void kernel_launch(void* const* d_in, const int* in_sizes, int n_in,
                              void* d_out, int out_size, void* d_ws, size_t ws_size,
                              hipStream_t stream) {
  const float* x     = (const float*)d_in[0];
  const int*   ei    = (const int*)d_in[1];
  const int*   src   = ei;
  const int*   dst   = ei + N_EDGES;
  const int*   batch = (const int*)d_in[2];
  const float* W[4]    = {(const float*)d_in[3], (const float*)d_in[6],
                          (const float*)d_in[9], (const float*)d_in[12]};
  const float* asrc[4] = {(const float*)d_in[4], (const float*)d_in[7],
                          (const float*)d_in[10], (const float*)d_in[13]};
  const float* adst[4] = {(const float*)d_in[5], (const float*)d_in[8],
                          (const float*)d_in[11], (const float*)d_in[14]};
  const float* mlp_w1 = (const float*)d_in[15];
  const float* mlp_b1 = (const float*)d_in[16];
  const float* gamma  = (const float*)d_in[17];
  const float* beta   = (const float*)d_in[18];
  const float* mlp_w2 = (const float*)d_in[19];
  const float* mlp_b2 = (const float*)d_in[20];

  // bump allocator over d_ws
  char* ws = (char*)d_ws;
  size_t off = 0;
  auto alloc = [&](size_t bytes) -> void* {
    void* p = ws + off;
    off = (off + bytes + 255) & ~(size_t)255;
    return p;
  };

  unsigned short* embb  = (unsigned short*)alloc((size_t)NPAD * K0PAD * 2);
  unsigned short* wt[4];
  wt[0] = (unsigned short*)alloc((size_t)512 * K0PAD * 2);
  for (int l = 1; l < 4; ++l) wt[l] = (unsigned short*)alloc((size_t)512 * HD * 2);
  unsigned short* wt_mlp = (unsigned short*)alloc((size_t)512 * 2048 * 2);
  float*          hproj = (float*)alloc((size_t)NPAD * HD * 4);
  float*          aggo  = (float*)alloc((size_t)NPAD * HD * 4);
  float*          h0    = (float*)alloc((size_t)NPAD * HD * 4);
  float*          x1    = (float*)alloc((size_t)NPAD * HD * 4);
  float*          x2    = (float*)alloc((size_t)NPAD * HD * 4);
  float*          x3    = (float*)alloc((size_t)NPAD * HD * 4);
  float*          xt    = (float*)alloc((size_t)NPAD * HD * 4);
  unsigned short* h0b   = (unsigned short*)alloc((size_t)NPAD * HD * 2);
  unsigned short* x1b   = (unsigned short*)alloc((size_t)NPAD * HD * 2);
  unsigned short* x3b   = (unsigned short*)alloc((size_t)NPAD * HD * 2);
  float*          es    = (float*)alloc((size_t)N_NODES * 4 * 4);
  float*          ed    = (float*)alloc((size_t)N_NODES * 4 * 4);
  unsigned*       menc  = (unsigned*)alloc((size_t)N_NODES * 4 * 4);
  float*          denom = (float*)alloc((size_t)N_NODES * 4 * 4);
  float*          ew    = (float*)alloc((size_t)N_EDGES * 4 * 4);
  unsigned*       penc  = (unsigned*)alloc((size_t)N_GRAPHS * 2048 * 4);
  float*          poolf = (float*)alloc((size_t)N_GRAPHS * 2048 * 4);
  unsigned short* poolb = (unsigned short*)alloc((size_t)N_GRAPHS * 2048 * 2);
  float*          ymat  = (float*)alloc((size_t)N_GRAPHS * HD * 4);

  // ---- embed + weight conversion (transposed bf16) ----
  embed_kernel<<<(NPAD + 255) / 256, 256, 0, stream>>>(x, embb);
  conv_wt_kernel<<<(512 * K0PAD + 255) / 256, 256, 0, stream>>>(W[0], wt[0], 39, K0PAD);
  for (int l = 1; l < 4; ++l)
    conv_wt_kernel<<<(512 * HD + 255) / 256, 256, 0, stream>>>(W[l], wt[l], HD, HD);
  conv_wt_kernel<<<(512 * 2048 + 255) / 256, 256, 0, stream>>>(mlp_w1, wt_mlp, 2048, 2048);

  const long nhd = (long)NPAD * HD;
  const int  gEH = (N_EDGES * 4 + 255) / 256;
  const int  gE  = (N_EDGES + 7) / 8;
  const int  gN  = (N_NODES + 7) / 8;
  const int  gHD = (int)((nhd + 255) / 256);

  auto gat_layer = [&](const unsigned short* Xbf, int Kin, int layer, float* out_buf,
                       unsigned short* out_bf, const float* resid, float* resid_out,
                       unsigned short* resid_bf) {
    gemm_bf16_wmma<<<dim3(HD / 128, NPAD / 128), 256, 0, stream>>>(
        Xbf, wt[layer], hproj, NPAD, Kin);
    scores_kernel<<<gN, 256, 0, stream>>>(hproj, asrc[layer], adst[layer], es, ed);
    fill_u32<<<(N_NODES * 4 + 255) / 256, 256, 0, stream>>>(menc, ENC_NEG_INF, N_NODES * 4);
    fill_u32<<<(N_NODES * 4 + 255) / 256, 256, 0, stream>>>((unsigned*)denom, 0u, N_NODES * 4);
    fill_u32<<<gHD, 256, 0, stream>>>((unsigned*)aggo, 0u, nhd);
    edge_logit_max<<<gEH, 256, 0, stream>>>(es, ed, src, dst, ew, menc);
    edge_w_denom<<<gEH, 256, 0, stream>>>(dst, ew, menc, denom);
    aggregate_kernel<<<gE, 256, 0, stream>>>(hproj, ew, denom, src, dst, aggo);
    elu_res_kernel<<<gHD, 256, 0, stream>>>(aggo, out_buf, out_bf, resid, resid_out,
                                            resid_bf, nhd);
  };

  gat_layer(embb, K0PAD, 0, h0, h0b, nullptr, nullptr, nullptr);  // init_graph
  gat_layer(h0b,  HD,    1, x1, x1b, nullptr, nullptr, nullptr);  // head
  gat_layer(x1b,  HD,    2, x2, nullptr, h0, x3, x3b);            // body (+res x3=h0+x2)
  gat_layer(x3b,  HD,    3, xt, nullptr, nullptr, nullptr, nullptr); // tail

  // ---- concat + segment_max pool ----
  fill_u32<<<(N_GRAPHS * 2048 + 255) / 256, 256, 0, stream>>>(penc, ENC_NEG_INF,
                                                              (long)N_GRAPHS * 2048);
  pool_max_kernel<<<(int)(((long)N_NODES * 2048 + 255) / 256), 256, 0, stream>>>(
      xt, x1, x2, x3, batch, penc);
  pool_dec_kernel<<<(N_GRAPHS * 2048 + 255) / 256, 256, 0, stream>>>(penc, poolf, poolb);

  // ---- MLP1 (WMMA) + BN/relu + final matvec ----
  gemm_bf16_wmma<<<dim3(HD / 128, N_GRAPHS / 128), 256, 0, stream>>>(
      poolb, wt_mlp, ymat, N_GRAPHS, 2048);
  bn_kernel<<<1, 512, 0, stream>>>(ymat, mlp_b1, gamma, beta);
  final_kernel<<<(N_GRAPHS + 7) / 8, 256, 0, stream>>>(ymat, mlp_w2, mlp_b2, (float*)d_out);
}
